// EquivariantGraphConvolution_45045617001163
// MI455X (gfx1250) — matrix operand order repaired and checked by reading, not compiled, not run on hardware
//
#include <hip/hip_runtime.h>
#include <hip/hip_bf16.h>

// ---------------------------------------------------------------------------
// EGNN layer for MI455X (gfx1250, wave32, WMMA).
// N=50000 nodes, E=800000 edges, D_H=128, D_E=16.
// Edge MLP:  [h_i|h_j|d2|e] (273) ->silu 128 ->silu 128 ; coord scalar via Wc.
// All GEMMs as v_wmma_f32_16x16x32_bf16 over 16-edge / 16-node tiles.
// ---------------------------------------------------------------------------

#define E_TOT 800000
#define N_TOT 50000
#define NFS   131            // node_features row stride (3 coords + 128 h)

typedef __attribute__((ext_vector_type(16))) __bf16         bf16x16;
typedef __attribute__((ext_vector_type(8)))  float          f32x8;
typedef __attribute__((ext_vector_type(8)))  unsigned short us8;

union Frag { bf16x16 bf; us8 h[2]; };

__device__ __forceinline__ unsigned short f2bf(float f) {
  unsigned u = __float_as_uint(f);
  u += 0x7FFFu + ((u >> 16) & 1u);        // round-to-nearest-even
  return (unsigned short)(u >> 16);
}
__device__ __forceinline__ float silu_f(float x) { return x / (1.0f + __expf(-x)); }
__device__ __forceinline__ void atomAddF(float* p, float v) { unsafeAtomicAdd(p, v); }
#define WAVE_LDS_SYNC() asm volatile("s_wait_dscnt 0x0" ::: "memory")

__device__ __forceinline__ f32x8 wmma_bf16(const Frag& a, const Frag& b, f32x8 c) {
  return __builtin_amdgcn_wmma_f32_16x16x32_bf16(false, a.bf, false, b.bf,
                                                 (short)0, c, false, false);
}

// Shared 16xK (A, LDS, row-major bf16) x Kx128 (B, packed global) -> 8 f32 accs.
// B packed layout: Bp[ntile][chunk][lane][elem], elem i of lane L = W[k][n],
//   n = ntile*16 + L%16,  k = chunk*32 + (L/16)*16 + i.
// A fragment (contiguous, matches B): lane L, row M=L%16,
//   elems 0..15 = k = c*32 + (L/16)*16 + 0..15
template<int KC, int ASTRIDE>
__device__ __forceinline__ void gemm16(const unsigned short* A,
                                       const unsigned short* __restrict__ Bp,
                                       int L, f32x8 acc[8]) {
  const int lm = L & 15, hi = L >> 4;
  const unsigned short* arow = A + lm * ASTRIDE + hi * 16;
#pragma unroll
  for (int c = 0; c < KC; ++c) {
    Frag a;
    a.h[0] = *(const us8*)(arow + c * 32);
    a.h[1] = *(const us8*)(arow + c * 32 + 8);
#pragma unroll
    for (int t = 0; t < 8; ++t) {
      Frag b;
      const unsigned short* bp = Bp + ((t * KC + c) << 9) + L * 16;
      b.h[0] = *(const us8*)bp;
      b.h[1] = *(const us8*)(bp + 8);
      acc[t] = wmma_bf16(a, b, acc[t]);
    }
  }
}

// ---------------- workspace zeroing ----------------
__global__ void egnn_zero(float* p, int n) {
  int i = blockIdx.x * blockDim.x + threadIdx.x;
  int stride = gridDim.x * blockDim.x;
  for (; i < n; i += stride) p[i] = 0.0f;
}

// ---------------- weight repack f32 -> bf16 B-layout ----------------
// regions: We1p 36864 (KC=9,K=273) | We2p 16384 (KC=4,K=128)
//        | Wn1p 32768 (KC=8,K=256) | Wn2p 16384 (KC=4,K=128)
__global__ void egnn_pack(const float* __restrict__ We1, const float* __restrict__ We2,
                          const float* __restrict__ Wn1, const float* __restrict__ Wn2,
                          unsigned short* We1p, unsigned short* We2p,
                          unsigned short* Wn1p, unsigned short* Wn2p) {
  int i = blockIdx.x * blockDim.x + threadIdx.x;
  const float* W; unsigned short* P; int KC, Kact, rel;
  if (i < 36864)       { W = We1; P = We1p; KC = 9; Kact = 273; rel = i; }
  else if (i < 53248)  { W = We2; P = We2p; KC = 4; Kact = 128; rel = i - 36864; }
  else if (i < 86016)  { W = Wn1; P = Wn1p; KC = 8; Kact = 256; rel = i - 53248; }
  else if (i < 102400) { W = Wn2; P = Wn2p; KC = 4; Kact = 128; rel = i - 86016; }
  else return;
  int nt   = rel / (KC * 512);
  int r2   = rel % (KC * 512);
  int c    = r2 / 512;
  int r3   = r2 % 512;
  int lane = r3 / 16, el = r3 % 16;
  int n = nt * 16 + (lane & 15);
  int k = c * 32 + (lane >> 4) * 16 + el;
  float v = (k < Kact) ? W[k * 128 + n] : 0.0f;
  P[rel] = f2bf(v);
}

// ---------------- fused edge kernel ----------------
// 4 waves/block, 16 edges per wave. Per-wave LDS slice:
//   [0,9216)      stage: 16 x 288 bf16 A tile   (later reused as 16x128 f32 m)
//   [9216,13312)  m1s:   16 x 128 bf16
//   [13312,13376) ss:    16 f32 coord scalars
#define EW_BYTES 13376
__global__ __launch_bounds__(128) void egnn_edge(
    const float* __restrict__ nf, const int* __restrict__ ei,
    const float* __restrict__ efeat,
    const unsigned short* __restrict__ We1p, const float* __restrict__ be1,
    const unsigned short* __restrict__ We2p, const float* __restrict__ be2,
    const float* __restrict__ Wc, const float* __restrict__ bc,
    float* __restrict__ nbr, float* __restrict__ seg, float* __restrict__ cnt) {
  __shared__ __align__(16) char smem[4 * EW_BYTES];
  const int w = threadIdx.x >> 5, L = threadIdx.x & 31;
  const int lm = L & 15, hi = L >> 4;
  char* sw = smem + w * EW_BYTES;
  unsigned short* stage = (unsigned short*)sw;
  unsigned short* m1s   = (unsigned short*)(sw + 9216);
  float*          ss    = (float*)(sw + 13312);
  float*          mf    = (float*)sw;          // reuse of stage after GEMM1

  const int  ebase  = (blockIdx.x * 4 + w) * 16;
  const int* first  = ei;
  const int* second = ei + E_TOT;

  // ---- stage A = [h_i | h_j | d2 | e | 0-pad] as 16 x 288 bf16 ----
  for (int e = 0; e < 16; ++e) {
    int eid = ebase + e;
    int fi = first[eid], si = second[eid];
    const float* hf = nf + (long)fi * NFS + 3;
    const float* hs = nf + (long)si * NFS + 3;
    for (int t = L; t < 128; t += 32) {
      stage[e * 288 + t]       = f2bf(hf[t]);
      stage[e * 288 + 128 + t] = f2bf(hs[t]);
    }
    if (L == 0) {
      float dx = nf[(long)fi * NFS + 0] - nf[(long)si * NFS + 0];
      float dy = nf[(long)fi * NFS + 1] - nf[(long)si * NFS + 1];
      float dz = nf[(long)fi * NFS + 2] - nf[(long)si * NFS + 2];
      stage[e * 288 + 256] = f2bf(dx * dx + dy * dy + dz * dz);
    }
    if (L < 16) stage[e * 288 + 257 + L] = f2bf(efeat[(long)eid * 16 + L]);
    if (L < 15) stage[e * 288 + 273 + L] = 0;
  }
  WAVE_LDS_SYNC();

  // ---- GEMM1: 16x288 @ 288x128, bias+silu, -> m1s bf16 ----
  f32x8 z = {0.f, 0.f, 0.f, 0.f, 0.f, 0.f, 0.f, 0.f};
  f32x8 acc[8];
#pragma unroll
  for (int t = 0; t < 8; ++t) acc[t] = z;
  gemm16<9, 288>(stage, We1p, L, acc);
#pragma unroll
  for (int t = 0; t < 8; ++t) {
    float bv = be1[t * 16 + lm];
#pragma unroll
    for (int r = 0; r < 8; ++r)
      m1s[(r + 8 * hi) * 128 + t * 16 + lm] = f2bf(silu_f(acc[t][r] + bv));
  }
  WAVE_LDS_SYNC();

  // ---- GEMM2: 16x128 @ 128x128 ----
  f32x8 acc2[8];
#pragma unroll
  for (int t = 0; t < 8; ++t) acc2[t] = z;
  gemm16<4, 128>(m1s, We2p, L, acc2);

  // ---- bias+silu -> m (f32, LDS), plus coord scalar m @ Wc ----
  float part[8];
#pragma unroll
  for (int r = 0; r < 8; ++r) part[r] = 0.0f;
#pragma unroll
  for (int t = 0; t < 8; ++t) {
    float bv = be2[t * 16 + lm];
    float wc = Wc[t * 16 + lm];
#pragma unroll
    for (int r = 0; r < 8; ++r) {
      float x = silu_f(acc2[t][r] + bv);
      mf[(r + 8 * hi) * 128 + t * 16 + lm] = x;
      part[r] += x * wc;
    }
  }
#pragma unroll
  for (int m = 1; m < 16; m <<= 1)
#pragma unroll
    for (int r = 0; r < 8; ++r) part[r] += __shfl_xor(part[r], m, 16);
  if (lm == 0) {
    float bcv = bc[0];
#pragma unroll
    for (int r = 0; r < 8; ++r) ss[r + 8 * hi] = part[r] + bcv;
  }
  WAVE_LDS_SYNC();

  // ---- segment scatters (L2-resident accumulators) ----
  for (int e = 0; e < 16; ++e) {
    int dst = first[ebase + e];
    float* np = nbr + (long)dst * 128;
    for (int t = L; t < 128; t += 32) atomAddF(np + t, mf[e * 128 + t]);
  }
  if (L < 16) {
    int eid = ebase + L;
    int fi = first[eid], si = second[eid];
    float s = ss[L];
#pragma unroll
    for (int j = 0; j < 3; ++j) {
      float d = nf[(long)fi * NFS + j] - nf[(long)si * NFS + j];
      atomAddF(seg + (long)fi * 3 + j, d * s);
    }
    atomAddF(cnt + fi, 1.0f);
  }
}

// ---------------- fused node kernel ----------------
// 4 waves/block, 16 nodes per wave. Per-wave LDS:
//   [0,8192) stage 16x256 bf16, [8192,12288) m1s 16x128 bf16
#define NW_BYTES 12288
__global__ __launch_bounds__(128) void egnn_node(
    const float* __restrict__ nf, const float* __restrict__ nbr,
    const float* __restrict__ seg, const float* __restrict__ cnt,
    const unsigned short* __restrict__ Wn1p, const float* __restrict__ bn1,
    const unsigned short* __restrict__ Wn2p, const float* __restrict__ bn2,
    float* __restrict__ out) {
  __shared__ __align__(16) char smem[4 * NW_BYTES];
  const int w = threadIdx.x >> 5, L = threadIdx.x & 31;
  const int lm = L & 15, hi = L >> 4;
  const int tile = blockIdx.x * 4 + w;
  if (tile >= N_TOT / 16) return;       // wave-uniform guard
  char* sw = smem + w * NW_BYTES;
  unsigned short* stage = (unsigned short*)sw;
  unsigned short* m1s   = (unsigned short*)(sw + 8192);
  const int nbase = tile * 16;

  // ---- stage A = [h | nbr_sum] as 16 x 256 bf16 ----
  for (int e = 0; e < 16; ++e) {
    int n = nbase + e;
    const float* hp = nf + (long)n * NFS + 3;
    const float* sp = nbr + (long)n * 128;
    for (int t = L; t < 128; t += 32) {
      stage[e * 256 + t]       = f2bf(hp[t]);
      stage[e * 256 + 128 + t] = f2bf(sp[t]);
    }
  }
  WAVE_LDS_SYNC();

  f32x8 z = {0.f, 0.f, 0.f, 0.f, 0.f, 0.f, 0.f, 0.f};
  f32x8 acc[8];
#pragma unroll
  for (int t = 0; t < 8; ++t) acc[t] = z;
  gemm16<8, 256>(stage, Wn1p, L, acc);
#pragma unroll
  for (int t = 0; t < 8; ++t) {
    float bv = bn1[t * 16 + lm];
#pragma unroll
    for (int r = 0; r < 8; ++r)
      m1s[(r + 8 * hi) * 128 + t * 16 + lm] = f2bf(silu_f(acc[t][r] + bv));
  }
  WAVE_LDS_SYNC();

  f32x8 acc2[8];
#pragma unroll
  for (int t = 0; t < 8; ++t) acc2[t] = z;
  gemm16<4, 128>(m1s, Wn2p, L, acc2);

  // ---- write packed output: nodes cols 3..130 ----
#pragma unroll
  for (int t = 0; t < 8; ++t) {
    float bv = bn2[t * 16 + lm];
#pragma unroll
    for (int r = 0; r < 8; ++r)
      out[(long)(nbase + r + 8 * hi) * NFS + 3 + t * 16 + lm] = acc2[t][r] + bv;
  }
  // ---- coords_new cols 0..2 ----
  if (L < 16) {
    int n = nbase + L;
    float den = fmaxf(cnt[n], 1.0f);
#pragma unroll
    for (int j = 0; j < 3; ++j)
      out[(long)n * NFS + j] = nf[(long)n * NFS + j] + seg[(long)n * 3 + j] / den;
  }
}

// ---------------------------------------------------------------------------
extern "C" void kernel_launch(void* const* d_in, const int* in_sizes, int n_in,
                              void* d_out, int out_size, void* d_ws, size_t ws_size,
                              hipStream_t stream) {
  const float* nf   = (const float*)d_in[0];
  const int*   ei   = (const int*)d_in[1];
  const float* ef   = (const float*)d_in[2];
  const float* We1  = (const float*)d_in[3];
  const float* be1  = (const float*)d_in[4];
  const float* We2  = (const float*)d_in[5];
  const float* be2  = (const float*)d_in[6];
  const float* Wc   = (const float*)d_in[7];
  const float* bc   = (const float*)d_in[8];
  const float* Wn1  = (const float*)d_in[9];
  const float* bn1  = (const float*)d_in[10];
  const float* Wn2  = (const float*)d_in[11];
  const float* bn2  = (const float*)d_in[12];

  // workspace layout (bytes), ~26.6 MB total, all 16B aligned
  char* ws = (char*)d_ws;
  float*          nbr  = (float*)(ws + 0);                 // 50000*128*4
  float*          seg  = (float*)(ws + 25600000);          // 50000*3*4
  float*          cnt  = (float*)(ws + 26200000);          // 50000*4
  unsigned short* We1p = (unsigned short*)(ws + 26400000); // 36864 bf16
  unsigned short* We2p = (unsigned short*)(ws + 26473728); // 16384 bf16
  unsigned short* Wn1p = (unsigned short*)(ws + 26506496); // 32768 bf16
  unsigned short* Wn2p = (unsigned short*)(ws + 26572032); // 16384 bf16

  egnn_zero<<<4096, 256, 0, stream>>>((float*)ws, 6600000);
  egnn_pack<<<400, 256, 0, stream>>>(We1, We2, Wn1, Wn2, We1p, We2p, Wn1p, Wn2p);
  egnn_edge<<<E_TOT / 64, 128, 0, stream>>>(nf, ei, ef, We1p, be1, We2p, be2,
                                            Wc, bc, nbr, seg, cnt);
  egnn_node<<<(N_TOT / 16 + 3) / 4, 128, 0, stream>>>(nf, nbr, seg, cnt,
                                                      Wn1p, bn1, Wn2p, bn2,
                                                      (float*)d_out);
}
